// CasualSelfAttention_21784074126160
// MI455X (gfx1250) — matrix operand order repaired and checked by reading, not compiled, not run on hardware
//
#include <hip/hip_runtime.h>
#include <hip/hip_bf16.h>

typedef __attribute__((ext_vector_type(16))) _Float16 v16h;
typedef __attribute__((ext_vector_type(8)))  float    v8f;

#define WMMA_F32_F16(a, b, c) \
  __builtin_amdgcn_wmma_f32_16x16x32_f16(false, (a), false, (b), (short)0, (c), false, false)

constexpr int BATCH  = 8;
constexpr int SEQ    = 1024;
constexpr int CH     = 768;
constexpr int NHEAD  = 12;
constexpr int HDIM   = 64;     // CH / NHEAD
constexpr int THREEC = 3 * CH; // 2304

// ---------------------------------------------------------------------------
// CDNA5 async global->LDS copy (16B per lane, tracked by ASYNCcnt).
// Generic pointers into LDS carry the LDS byte offset in their low 32 bits
// so truncation yields the
// VDST address the instruction wants.
// ---------------------------------------------------------------------------
__device__ __forceinline__ void async_ld_b128(void* lds_ptr, const void* gptr) {
  unsigned int lds_off = (unsigned int)(unsigned long long)lds_ptr;
  unsigned long long ga = (unsigned long long)gptr;
  asm volatile("global_load_async_to_lds_b128 %0, %1, off"
               :: "v"(lds_off), "v"(ga) : "memory");
}
__device__ __forceinline__ void wait_async0() {
  asm volatile("s_wait_asynccnt 0x0" ::: "memory");
}

// ---------------------------------------------------------------------------
// 16-lane reductions (xor masks 1/2/4/8 stay inside each 16-lane half of the
// wave, matching the WMMA C/D layout's M-split).
// ---------------------------------------------------------------------------
__device__ __forceinline__ float rowmax16(float v) {
  v = fmaxf(v, __shfl_xor(v, 1));
  v = fmaxf(v, __shfl_xor(v, 2));
  v = fmaxf(v, __shfl_xor(v, 4));
  v = fmaxf(v, __shfl_xor(v, 8));
  return v;
}
__device__ __forceinline__ float rowsum16(float v) {
  v += __shfl_xor(v, 1);
  v += __shfl_xor(v, 2);
  v += __shfl_xor(v, 4);
  v += __shfl_xor(v, 8);
  return v;
}

// ---------------------------------------------------------------------------
// WMMA GEMM: Out[M,N] = A[M,K] * W[K,N] + bias[N]
// Block 256 threads = 8 waves; block tile 64(M) x 128(N); wave grid 2(M)x4(N);
// each wave owns a 32x32 strip: 2 A-frags x 2 B-frags -> 4 WMMAs per K-step
// (K-tile 32), so LDS gather traffic per WMMA is halved vs a 16x32 strip.
//   A_F16 : A is f16 -> staged with async global->LDS b128 copies
//   SCALE_Q: cols < CH scaled by 1/sqrt(HDIM)=0.125 (folds attn scale into Q)
//   OUT_F16: store f16 (else f32)
// ---------------------------------------------------------------------------
template <bool A_F16, bool SCALE_Q, bool OUT_F16>
__global__ __launch_bounds__(256) void gemm_bias_wmma(
    const void* __restrict__ Ap, const float* __restrict__ Wp,
    const float* __restrict__ bias, void* __restrict__ Outp,
    int M, int N, int K) {
  __shared__ alignas(16) _Float16 As[64][40];   // [m][k]  (stride 80B, 16B-aligned rows)
  __shared__ alignas(16) _Float16 Bs[128][40];  // [n][k]  (W transposed)

  const int tid    = threadIdx.x;
  const int lane   = tid & 31;
  const int w      = tid >> 5;
  const int half16 = lane >> 4;
  const int l16    = lane & 15;
  const int waveM  = w & 1;         // 2 M-groups of 32 rows
  const int waveN  = w >> 1;        // 4 N-groups of 32 cols
  const int rm     = waveM * 32;
  const int rn     = waveN * 32;
  const int m0     = blockIdx.x * 64;
  const int n0     = blockIdx.y * 128;

  // cooperative-load mapping
  const int ar = tid >> 2, as_ = (tid & 3);       // A: 64 rows, 4 x 8-col segments
  const int bk = tid >> 3, bn = (tid & 7) * 16;   // W: 32 rows x 128 cols, 16/thread

  v8f acc[2][2];
  const v8f vzero = {};
#pragma unroll
  for (int mt = 0; mt < 2; ++mt)
#pragma unroll
    for (int nt = 0; nt < 2; ++nt) acc[mt][nt] = vzero;

  for (int k0 = 0; k0 < K; k0 += 32) {
    __syncthreads();
    if constexpr (A_F16) {
      // straight f16 copy: async DMA into LDS, no VGPR round-trip
      const _Float16* a = (const _Float16*)Ap + (size_t)(m0 + ar) * K + k0 + as_ * 8;
      async_ld_b128(&As[ar][as_ * 8], a);
    } else {
      const float* a = (const float*)Ap + (size_t)(m0 + ar) * K + k0 + as_ * 8;
#pragma unroll
      for (int j = 0; j < 8; ++j) As[ar][as_ * 8 + j] = (_Float16)a[j];
    }
    {
      const float* wp = Wp + (size_t)(k0 + bk) * N + n0 + bn;
#pragma unroll
      for (int j = 0; j < 16; ++j) Bs[bn + j][bk] = (_Float16)wp[j];
    }
    if constexpr (A_F16) wait_async0();
    __syncthreads();

    // A fragments, ISA 16-bit 16x32 layout:
    // lanes 0-15: K = {0..7,16..23}; lanes 16-31: K = {8..15,24..31}
    v16h af[2];
#pragma unroll
    for (int mt = 0; mt < 2; ++mt) {
      const _Float16* ap = &As[rm + mt * 16 + l16][8 * half16];
#pragma unroll
      for (int i = 0; i < 4; ++i) {
        af[mt][2 * i]     = ap[2 * i];
        af[mt][2 * i + 1] = ap[2 * i + 1];
        af[mt][8 + 2 * i] = ap[16 + 2 * i];
        af[mt][9 + 2 * i] = ap[16 + 2 * i + 1];
      }
    }
    // B fragments: lane = column N; same K order as A
    // (lanes 0-15: K = {0..7,16..23}; lanes 16-31: K = {8..15,24..31})
    v16h bf[2];
#pragma unroll
    for (int nt = 0; nt < 2; ++nt) {
      const _Float16* bp = &Bs[rn + nt * 16 + l16][8 * half16];
#pragma unroll
      for (int i = 0; i < 16; ++i) bf[nt][i] = bp[(i & 7) + 16 * (i >> 3)];
    }
#pragma unroll
    for (int mt = 0; mt < 2; ++mt)
#pragma unroll
      for (int nt = 0; nt < 2; ++nt)
        acc[mt][nt] = WMMA_F32_F16(af[mt], bf[nt], acc[mt][nt]);
  }

  // C/D layout: VGPR r holds (M = r + 8*half16, N = l16)
#pragma unroll
  for (int mt = 0; mt < 2; ++mt)
#pragma unroll
    for (int nt = 0; nt < 2; ++nt) {
      const int col  = n0 + rn + nt * 16 + l16;
      const float bv = bias[col];
#pragma unroll
      for (int r = 0; r < 8; ++r) {
        const int row = m0 + rm + mt * 16 + r + 8 * half16;
        float v = acc[mt][nt][r] + bv;
        if (SCALE_Q && col < CH) v *= 0.125f;  // 1/sqrt(64) folded into Q
        if (OUT_F16) ((_Float16*)Outp)[(size_t)row * N + col] = (_Float16)v;
        else         ((float*)Outp)[(size_t)row * N + col]    = v;
      }
    }
}

// ---------------------------------------------------------------------------
// Flash attention (causal), f16 operands / f32 accum.
// Grid: (B*NH, SEQ/128). Block 256 = 8 waves; wave w owns q rows
// [blockIdx.y*128 + w*16, +16). Key tiles of 32. K tile is a straight f16
// copy -> async global->LDS; V tile is staged transposed (cooperative) so
// P@V B-fragments gather contiguously. Scale already folded into stored Q.
// ---------------------------------------------------------------------------
__global__ __launch_bounds__(256) void flash_attn_wmma(
    const _Float16* __restrict__ qkv, _Float16* __restrict__ y) {
  __shared__ alignas(16) _Float16 Ks[32][72];    // [key][d] (stride 144B)
  __shared__ alignas(16) _Float16 VTs[64][36];   // [d][key]
  __shared__ alignas(16) _Float16 Ps[8][16][34]; // per-wave P tile staging

  const int tid    = threadIdx.x;
  const int lane   = tid & 31;
  const int w      = tid >> 5;
  const int half16 = lane >> 4;
  const int l16    = lane & 15;
  const int bh     = blockIdx.x;
  const int b      = bh / NHEAD;
  const int h      = bh % NHEAD;
  const int q0     = blockIdx.y * 128 + w * 16;

  const _Float16* qbase = qkv + (size_t)b * SEQ * THREEC + h * HDIM;

  // resident Q fragments (2 K-steps of 32 over HDIM=64)
  v16h qf[2];
  {
    const _Float16* qp = qbase + (size_t)(q0 + l16) * THREEC;
#pragma unroll
    for (int ds = 0; ds < 2; ++ds) {
      const _Float16* p = qp + ds * 32 + 8 * half16;
#pragma unroll
      for (int i = 0; i < 4; ++i) {
        qf[ds][2 * i]     = p[2 * i];
        qf[ds][2 * i + 1] = p[2 * i + 1];
        qf[ds][8 + 2 * i] = p[16 + 2 * i];
        qf[ds][9 + 2 * i] = p[16 + 2 * i + 1];
      }
    }
  }

  const v8f vzero = {};
  float m[8], l[8];
  v8f o[4];
#pragma unroll
  for (int r = 0; r < 8; ++r) { m[r] = -3.0e38f; l[r] = 0.0f; }
#pragma unroll
  for (int nt = 0; nt < 4; ++nt) o[nt] = vzero;

  const int ck = tid >> 3;        // key 0..31
  const int cd = (tid & 7) * 8;   // d segment base

  const int ntiles = blockIdx.y * 4 + 4;  // keys up to end of this q block
  for (int kt = 0; kt < ntiles; ++kt) {
    const int k0 = kt * 32;
    __syncthreads();
    {
      // K tile: straight f16 copy -> async DMA (ASYNCcnt)
      const _Float16* kp = qbase + CH + (size_t)(k0 + ck) * THREEC + cd;
      async_ld_b128(&Ks[ck][cd], kp);
      // V tile: transposed store (must go through VGPRs)
      const _Float16* vp = qbase + 2 * CH + (size_t)(k0 + ck) * THREEC + cd;
#pragma unroll
      for (int j = 0; j < 8; ++j) VTs[cd + j][ck] = vp[j];
    }
    wait_async0();
    __syncthreads();

    if (k0 > q0 + 15) continue;  // wave-uniform: EXEC stays full around WMMA

    // S = Q @ K^T for 16 q-rows x 32 keys
    v8f s0 = vzero, s1 = vzero;
#pragma unroll
    for (int ds = 0; ds < 2; ++ds) {
      v16h b0, b1;
      const _Float16* bp0 = &Ks[l16][ds * 32 + 8 * half16];
      const _Float16* bp1 = &Ks[16 + l16][ds * 32 + 8 * half16];
#pragma unroll
      for (int i = 0; i < 16; ++i) { b0[i] = bp0[(i & 7) + 16 * (i >> 3)]; b1[i] = bp1[(i & 7) + 16 * (i >> 3)]; }
      s0 = WMMA_F32_F16(qf[ds], b0, s0);
      s1 = WMMA_F32_F16(qf[ds], b1, s1);
    }

    // causal mask on accumulator lanes
#pragma unroll
    for (int r = 0; r < 8; ++r) {
      const int row = q0 + r + 8 * half16;
      if (k0 + l16 > row)      s0[r] = -3.0e38f;
      if (k0 + 16 + l16 > row) s1[r] = -3.0e38f;
    }

    // online softmax update + stage P (f16) in wave-private LDS (C-layout)
#pragma unroll
    for (int r = 0; r < 8; ++r) {
      const float tmax  = rowmax16(fmaxf(s0[r], s1[r]));
      const float mn    = fmaxf(m[r], tmax);
      const float alpha = exp2f((m[r] - mn) * 1.44269504f);
      m[r] = mn;
      const float p0 = exp2f((s0[r] - mn) * 1.44269504f);
      const float p1 = exp2f((s1[r] - mn) * 1.44269504f);
      l[r] = l[r] * alpha + rowsum16(p0 + p1);
#pragma unroll
      for (int nt = 0; nt < 4; ++nt) o[nt][r] *= alpha;
      Ps[w][r + 8 * half16][l16]      = (_Float16)p0;
      Ps[w][r + 8 * half16][16 + l16] = (_Float16)p1;
    }

    // O += P @ V  (A-fragment re-gather of P, contiguous V^T B-fragments)
    v16h pf;
    {
      const _Float16* pp = &Ps[w][l16][8 * half16];
#pragma unroll
      for (int i = 0; i < 4; ++i) {
        pf[2 * i]     = pp[2 * i];
        pf[2 * i + 1] = pp[2 * i + 1];
        pf[8 + 2 * i] = pp[16 + 2 * i];
        pf[9 + 2 * i] = pp[16 + 2 * i + 1];
      }
    }
#pragma unroll
    for (int nt = 0; nt < 4; ++nt) {
      v16h bf;
      const _Float16* vp2 = &VTs[nt * 16 + l16][8 * half16];
#pragma unroll
      for (int i = 0; i < 16; ++i) bf[i] = vp2[(i & 7) + 16 * (i >> 3)];
      o[nt] = WMMA_F32_F16(pf, bf, o[nt]);
    }
  }

  // normalize and store y[b, t, h*64 + d] as f16 for the projection GEMM
#pragma unroll
  for (int r = 0; r < 8; ++r) {
    const float inv = 1.0f / l[r];
    const int trow  = q0 + r + 8 * half16;
    _Float16* yp = y + (size_t)(b * SEQ + trow) * CH + h * HDIM;
#pragma unroll
    for (int nt = 0; nt < 4; ++nt)
      yp[nt * 16 + l16] = (_Float16)(o[nt][r] * inv);
  }
}

// ---------------------------------------------------------------------------
// launch: qkv-GEMM -> flash attention -> projection GEMM
// ---------------------------------------------------------------------------
extern "C" void kernel_launch(void* const* d_in, const int* in_sizes, int n_in,
                              void* d_out, int out_size, void* d_ws, size_t ws_size,
                              hipStream_t stream) {
  const float* x      = (const float*)d_in[0];
  const float* W_attn = (const float*)d_in[1];
  const float* b_attn = (const float*)d_in[2];
  const float* W_proj = (const float*)d_in[3];
  const float* b_proj = (const float*)d_in[4];
  float* out = (float*)d_out;

  _Float16* qkv = (_Float16*)d_ws;                       // [8192, 2304] f16
  _Float16* yh  = qkv + (size_t)BATCH * SEQ * THREEC;    // [8192, 768]  f16

  const int M = BATCH * SEQ;  // 8192

  // qkv = x @ W_attn + b_attn (q scaled by 0.125, stored f16)
  gemm_bias_wmma<false, true, true>
      <<<dim3(M / 64, THREEC / 128), 256, 0, stream>>>(x, W_attn, b_attn, qkv,
                                                       M, THREEC, CH);
  // causal flash attention, y stored f16
  flash_attn_wmma<<<dim3(BATCH * NHEAD, SEQ / 128), 256, 0, stream>>>(qkv, yh);
  // out = y @ W_proj + b_proj (f32 output)
  gemm_bias_wmma<true, false, false>
      <<<dim3(M / 64, CH / 128), 256, 0, stream>>>(yh, W_proj, b_proj, out,
                                                   M, CH, CH);
}
